// TemporalSkeletonBranch_10402410791072
// MI455X (gfx1250) — matrix operand order repaired and checked by reading, not compiled, not run on hardware
//
#include <hip/hip_runtime.h>
#include <hip/hip_bf16.h>

typedef unsigned int u32;
typedef unsigned short u16;
typedef __attribute__((ext_vector_type(16))) __bf16 v16bf;
typedef __attribute__((ext_vector_type(2)))  __bf16 v2bf;
typedef __attribute__((ext_vector_type(8)))  float  v8f;
typedef __attribute__((ext_vector_type(4)))  u32    u32x4;
typedef __attribute__((ext_vector_type(8)))  int    i32x8;
typedef __attribute__((ext_vector_type(4)))  int    i32x4;

#define HID 128
#define JP 48             // joints padded 33 -> 48
#define SAMP 2            // samples per workgroup
#define MROWS (SAMP*JP)   // 96

// ---------------- workspace layout (bytes) ----------------
#define OFF_W0T   0u          // u16 [128][64]
#define OFF_S0T   16384u      // u16 [128][64]
#define OFF_W1T   32768u      // u16 [128][128]
#define OFF_W2T   65536u      // u16 [128][128]
#define OFF_TW1T  98304u      // u16 [128][128]
#define OFF_ADJP  131072u     // u16 [48][64]
#define OFF_PAR   137216u     // float[1280]: sb0,b0,sc0,sh0,b1,sc1,sh1,b2,sc2,sh2
#define OFF_XT    147456u     // float[16384][128]

#if defined(__gfx1250__) && __has_builtin(__builtin_amdgcn_tensor_load_to_lds) && \
    __has_builtin(__builtin_amdgcn_s_wait_tensorcnt)
#define HAVE_TDM 1
#endif

// ---- f32 -> bf16 conversions ----
// gfx1250 has native packed convert; clang doesn't declare the builtin on this
// toolchain, so emit it via inline asm (RNE, one VALU op per pair).
#if defined(__gfx1250__)
__device__ __forceinline__ u32 f2bf_pk(float a, float b) {
  u32 r;
  asm("v_cvt_pk_bf16_f32 %0, %1, %2" : "=v"(r) : "v"(a), "v"(b));
  return r;
}
__device__ __forceinline__ u16 f2bf(float f) {
  u32 r;
  asm("v_cvt_pk_bf16_f32 %0, %1, %1" : "=v"(r) : "v"(f));
  return (u16)r;
}
#else
__device__ __forceinline__ u16 f2bf(float f) {
  u32 u = __float_as_uint(f);
  u32 r = (u + 0x7FFFu + ((u >> 16) & 1u)) >> 16;
  return (u16)r;
}
__device__ __forceinline__ u32 f2bf_pk(float a, float b) {
  return (u32)f2bf(a) | ((u32)f2bf(b) << 16);
}
#endif

__device__ __forceinline__ float bf2f(u16 h) {
  return __uint_as_float(((u32)h) << 16);
}
__device__ __forceinline__ float gelu_exact(float v) {
  return 0.5f * v * (1.0f + erff(v * 0.70710678118654752f));
}

union ABFrag { v16bf v; uint4 q[2]; u16 u[16]; };

// A fragment (16x32 bf16) from row-major [.. , stride] buffer.
__device__ __forceinline__ void load_afrag(ABFrag& a, const u16* base, int stride,
                                           int row, int k0, int lane) {
  int ka = (lane & 16) ? 8 : 0;
  const u16* p = base + row * stride + k0 + ka;
  a.q[0] = *(const uint4*)(p);
  a.q[1] = *(const uint4*)(p + 16);
}
// B fragment (32x16 bf16) from K-contiguous (transposed) [.. , stride] buffer.
__device__ __forceinline__ void load_bfrag(ABFrag& b, const u16* base, int stride,
                                           int col, int k0, int lane) {
  int kb = (lane & 16) ? 8 : 0;
  const u16* p = base + col * stride + k0 + kb;
  b.q[0] = *(const uint4*)(p);
  b.q[1] = *(const uint4*)(p + 16);
}

#define WMMA_BF16(A, B, C) \
  __builtin_amdgcn_wmma_f32_16x16x32_bf16(false, (A), false, (B), (short)0, (C), false, false)

#define V8F_ZERO {0.f,0.f,0.f,0.f,0.f,0.f,0.f,0.f}

// pack 8 accumulator floats (one D tile slice) to bf16 and store 16B into tT
__device__ __forceinline__ void store_tile_tT(u16* tT, int c, int base_row, const v8f& acc) {
  u32 q[4];
  q[0] = f2bf_pk(acc[0], acc[1]);
  q[1] = f2bf_pk(acc[2], acc[3]);
  q[2] = f2bf_pk(acc[4], acc[5]);
  q[3] = f2bf_pk(acc[6], acc[7]);
  *(uint4*)&tT[c * 128 + base_row] = *(const uint4*)q;
}

// ================= kernel 1: pack weights / params =================
struct PackArgs { const float* p[26]; };

__global__ void pack_kernel(PackArgs in, char* ws) {
  const float* adj = in.p[1];
  const float* W0 = in.p[2];
  const float* S0 = in.p[4];  const float* sb0 = in.p[5];
  const float* g0 = in.p[6];  const float* be0 = in.p[7];
  const float* m0 = in.p[8];  const float* v0 = in.p[9];
  const float* W1 = in.p[10]; const float* b1 = in.p[11];
  const float* g1 = in.p[12]; const float* be1 = in.p[13];
  const float* m1 = in.p[14]; const float* v1 = in.p[15];
  const float* W2 = in.p[16]; const float* b2 = in.p[17];
  const float* g2 = in.p[18]; const float* be2 = in.p[19];
  const float* m2 = in.p[20]; const float* v2 = in.p[21];
  const float* tW1 = in.p[22];
  const float* b0 = in.p[3];

  u16* W0t  = (u16*)(ws + OFF_W0T);
  u16* S0t  = (u16*)(ws + OFF_S0T);
  u16* W1t  = (u16*)(ws + OFF_W1T);
  u16* W2t  = (u16*)(ws + OFF_W2T);
  u16* tW1t = (u16*)(ws + OFF_TW1T);
  u16* adjP = (u16*)(ws + OFF_ADJP);
  float* par = (float*)(ws + OFF_PAR);

  int stride = gridDim.x * blockDim.x;
  for (int i = blockIdx.x * blockDim.x + threadIdx.x; i < HID * HID; i += stride) {
    int c = i >> 7;      // output col 0..127
    int k = i & 127;     // input k 0..127
    W1t[c * 128 + k]  = f2bf(W1[k * 128 + c]);
    W2t[c * 128 + k]  = f2bf(W2[k * 128 + c]);
    tW1t[c * 128 + k] = f2bf(tW1[k * 128 + c]);
    if (k < 64) {
      W0t[c * 64 + k] = f2bf(k < 52 ? W0[k * 128 + c] : 0.0f);
      S0t[c * 64 + k] = f2bf(k < 52 ? S0[k * 128 + c] : 0.0f);
    }
    if (c < JP && k < 64) {
      adjP[c * 64 + k] = f2bf((c < 33 && k < 33) ? adj[c * 33 + k] : 0.0f);
    }
    if (i < HID) {
      float s0v = g0[i] * rsqrtf(v0[i] + 1e-5f);
      float s1v = g1[i] * rsqrtf(v1[i] + 1e-5f);
      float s2v = g2[i] * rsqrtf(v2[i] + 1e-5f);
      par[0 * 128 + i] = sb0[i];
      par[1 * 128 + i] = b0[i];
      par[2 * 128 + i] = s0v;
      par[3 * 128 + i] = be0[i] - m0[i] * s0v;
      par[4 * 128 + i] = b1[i];
      par[5 * 128 + i] = s1v;
      par[6 * 128 + i] = be1[i] - m1[i] * s1v;
      par[7 * 128 + i] = b2[i];
      par[8 * 128 + i] = s2v;
      par[9 * 128 + i] = be2[i] - m2[i] * s2v;
    }
  }
}

// ================= kernel 2: fused 3-layer GCN =================
// LDS: hbuf u16[96][128], tT u16[128][128], adjbuf u16[48][64]

__device__ __forceinline__ void gemm_hW(const u16* hbuf, u16* tT, const u16* Wt,
                                        int Kdim, int w, int lane) {
  int c = w * 16 + (lane & 15);
  int hi8 = (lane & 16) ? 8 : 0;
  v8f acc[6] = {V8F_ZERO, V8F_ZERO, V8F_ZERO, V8F_ZERO, V8F_ZERO, V8F_ZERO};
  for (int k0 = 0; k0 < Kdim; k0 += 32) {
    ABFrag b; load_bfrag(b, Wt, Kdim, c, k0, lane);
    for (int rt = 0; rt < 6; rt++) {
      ABFrag a; load_afrag(a, hbuf, HID, rt * 16 + (lane & 15), k0, lane);
      acc[rt] = WMMA_BF16(a.v, b.v, acc[rt]);
    }
  }
  for (int rt = 0; rt < 6; rt++)
    store_tile_tT(tT, c, rt * 16 + hi8, acc[rt]);
}

__device__ __forceinline__ void gemm_hW0_dual(const u16* hbuf, u16* tT,
                                              const u16* W0t, const u16* S0t,
                                              int w, int lane, v8f skipacc[6]) {
  int c = w * 16 + (lane & 15);
  int hi8 = (lane & 16) ? 8 : 0;
  v8f acc[6] = {V8F_ZERO, V8F_ZERO, V8F_ZERO, V8F_ZERO, V8F_ZERO, V8F_ZERO};
  for (int rt = 0; rt < 6; rt++) skipacc[rt] = (v8f)V8F_ZERO;
  for (int k0 = 0; k0 < 64; k0 += 32) {
    ABFrag bw; load_bfrag(bw, W0t, 64, c, k0, lane);
    ABFrag bs; load_bfrag(bs, S0t, 64, c, k0, lane);
    for (int rt = 0; rt < 6; rt++) {
      ABFrag a; load_afrag(a, hbuf, HID, rt * 16 + (lane & 15), k0, lane);
      acc[rt]     = WMMA_BF16(a.v, bw.v, acc[rt]);
      skipacc[rt] = WMMA_BF16(a.v, bs.v, skipacc[rt]);
    }
  }
  for (int rt = 0; rt < 6; rt++)
    store_tile_tT(tT, c, rt * 16 + hi8, acc[rt]);
}

// z = adjbuf @ t; epilogue: +bias, BN affine, GeLU, +skip(hbuf), write back to hbuf
__device__ __forceinline__ void gemm_adj_ep(u16* hbuf, const u16* tT, const u16* adjbuf,
                                            const float* par, int layer_par,
                                            int w, int lane) {
  int c = w * 16 + (lane & 15);
  int hi8 = (lane & 16) ? 8 : 0;
  float biasc = par[(layer_par + 0) * 128 + c];
  float scc   = par[(layer_par + 1) * 128 + c];
  float shc   = par[(layer_par + 2) * 128 + c];
  for (int s = 0; s < SAMP; s++) {
    v8f acc[3] = {V8F_ZERO, V8F_ZERO, V8F_ZERO};
    for (int k0 = 0; k0 < 64; k0 += 32) {
      ABFrag b; load_bfrag(b, tT, 128, c, s * JP + k0, lane);
      for (int mt = 0; mt < 3; mt++) {
        ABFrag a; load_afrag(a, adjbuf, 64, mt * 16 + (lane & 15), k0, lane);
        acc[mt] = WMMA_BF16(a.v, b.v, acc[mt]);
      }
    }
    for (int mt = 0; mt < 3; mt++) {
#pragma unroll
      for (int r = 0; r < 8; r++) {
        int row = s * JP + mt * 16 + r + hi8;
        float v = acc[mt][r] + biasc;
        v = v * scc + shc;
        float g = gelu_exact(v);
        float skip = bf2f(hbuf[row * HID + c]);
        hbuf[row * HID + c] = f2bf(g + skip);
      }
    }
  }
}

__global__ __launch_bounds__(256)
void gcn_kernel(const float* __restrict__ x, const char* __restrict__ ws_c,
                float* __restrict__ xt) {
  extern __shared__ char smem[];
  u16* hbuf   = (u16*)smem;                                     // [96][128]
  u16* tT     = (u16*)(smem + MROWS * HID * 2);                 // [128][128]
  u16* adjbuf = (u16*)(smem + MROWS * HID * 2 + 128 * 128 * 2); // [48][64]

  const u16* W0t  = (const u16*)(ws_c + OFF_W0T);
  const u16* S0t  = (const u16*)(ws_c + OFF_S0T);
  const u16* W1t  = (const u16*)(ws_c + OFF_W1T);
  const u16* W2t  = (const u16*)(ws_c + OFF_W2T);
  const u16* adjPg = (const u16*)(ws_c + OFF_ADJP);
  const float* par = (const float*)(ws_c + OFF_PAR);

  const int tid = threadIdx.x;
  const int w = tid >> 5, lane = tid & 31;
  const int n0 = blockIdx.x * SAMP;

  // zero LDS (pad rows/cols must be 0 so adjP zero-columns guarantee correctness)
  for (int i = tid; i < MROWS * HID; i += 256) hbuf[i] = 0;
  for (int i = tid; i < 128 * 128; i += 256) tT[i] = 0;
  __syncthreads();

  // ---- stage adjP (48x64 bf16, 6KB) into LDS via Tensor Data Mover ----
#if defined(HAVE_TDM)
  if (tid < 32) {
    u32 lds_off = (u32)(uintptr_t)adjbuf;   // generic->LDS: low 32 bits are LDS offset
    unsigned long long ga = (unsigned long long)(uintptr_t)adjPg;
    // D# group0: count=1, lds_addr, global_addr[56:0], type=2
    u32x4 g0 = { 1u, lds_off, (u32)ga,
                 (u32)((ga >> 32) & 0x01FFFFFFu) | 0x80000000u };
    // D# group1: data_size=2B; 1-D tensor/tile of 3072 elements; dim0_stride=3072
    i32x8 g1 = { (int)0x00010000,   // workgroup_mask=0, data_size=1(2B)
                 (int)0x0C000000,   // tensor_dim0 low16 (3072) << 16
                 (int)0x00010000,   // tensor_dim0 hi16=0 | tensor_dim1=1 << 16
                 (int)0x0C000000,   // tensor_dim1 hi16=0 | tile_dim0 (3072) << 16
                 0,                 // tile_dim1=0, tile_dim2=0
                 3072,              // tensor_dim0_stride low32
                 0, 0 };
    i32x4 gz4 = { 0, 0, 0, 0 };
    i32x8 gz8 = { 0, 0, 0, 0, 0, 0, 0, 0 };
    __builtin_amdgcn_tensor_load_to_lds(g0, g1, gz4, gz4, gz8, 0);
    __builtin_amdgcn_s_wait_tensorcnt(0);   // DMA complete before barrier release
  }
#else
  for (int i = tid; i < JP * 64; i += 256) adjbuf[i] = adjPg[i];
#endif

  // stage x -> hbuf (bf16), rows s*48+j, cols 0..51 (52..63 stay zero)
  for (int idx = tid; idx < SAMP * 33 * 52; idx += 256) {
    int s = idx / (33 * 52);
    int r = idx - s * (33 * 52);
    int j = r / 52, f = r - j * 52;
    float val = x[((long)(n0 + s) * 33 + j) * 52 + f];
    hbuf[(s * JP + j) * HID + f] = f2bf(val);
  }
  __builtin_prefetch(W0t, 0, 1);
  __builtin_prefetch(W1t, 0, 1);
  __syncthreads();

  // ---- layer 0 ----
  v8f skipacc[6];
  gemm_hW0_dual(hbuf, tT, W0t, S0t, w, lane, skipacc);
  __syncthreads();                       // all waves done reading x from hbuf
  {
    int c = w * 16 + (lane & 15);
    int hi8 = (lane & 16) ? 8 : 0;
    float sb = par[0 * 128 + c];         // sb0
    for (int rt = 0; rt < 6; rt++)
#pragma unroll
      for (int r = 0; r < 8; r++)
        hbuf[(rt * 16 + r + hi8) * HID + c] = f2bf(skipacc[rt][r] + sb);
  }
  __syncthreads();
  gemm_adj_ep(hbuf, tT, adjbuf, par, 1, w, lane);
  __syncthreads();

  // ---- layer 1 ----
  gemm_hW(hbuf, tT, W1t, 128, w, lane);
  __syncthreads();
  gemm_adj_ep(hbuf, tT, adjbuf, par, 4, w, lane);
  __syncthreads();

  // ---- layer 2 ----
  gemm_hW(hbuf, tT, W2t, 128, w, lane);
  __syncthreads();
  gemm_adj_ep(hbuf, tT, adjbuf, par, 7, w, lane);
  __syncthreads();

  // joint mean -> xt[n][c]
  {
    int s = tid >> 7, c = tid & 127;
    float sum = 0.0f;
    for (int j = 0; j < 33; j++) sum += bf2f(hbuf[(s * JP + j) * HID + c]);
    xt[(long)(n0 + s) * HID + c] = sum * (1.0f / 33.0f);
  }
}

// ================= kernel 3: temporal attention =================
__global__ __launch_bounds__(256)
void attn_kernel(const float* __restrict__ xt, const char* __restrict__ ws_c,
                 const float* __restrict__ tb1, const float* __restrict__ tW2,
                 const float* __restrict__ tb2, float* __restrict__ out) {
  extern __shared__ char smem[];
  u16* xbuf = (u16*)smem;                        // [256][128] bf16
  float* score = (float*)(smem + 256 * 128 * 2); // [256]
  float* red   = score + 256;                    // [256]

  const u16* tW1t = (const u16*)(ws_c + OFF_TW1T);
  const int tid = threadIdx.x;
  const int w = tid >> 5, lane = tid & 31;
  const long base = (long)blockIdx.x * 256 * HID;

  for (int i = tid; i < (256 * HID) / 2; i += 256) {
    u32 pk = f2bf_pk(xt[base + i * 2 + 0], xt[base + i * 2 + 1]);
    *(u32*)&xbuf[i * 2] = pk;
  }
  score[tid] = tb2[0];
  __syncthreads();

  int c = w * 16 + (lane & 15);
  int hi8 = (lane & 16) ? 8 : 0;
  float tb1c = tb1[c], tw2c = tW2[c];

  // hoist all 4 k-step B fragments of tW1 (reused by all 16 M tiles)
  ABFrag bf[4];
#pragma unroll
  for (int kk = 0; kk < 4; kk++) load_bfrag(bf[kk], tW1t, 128, c, kk * 32, lane);

  for (int mt = 0; mt < 16; mt++) {
    v8f acc = V8F_ZERO;
#pragma unroll
    for (int kk = 0; kk < 4; kk++) {
      ABFrag a; load_afrag(a, xbuf, HID, mt * 16 + (lane & 15), kk * 32, lane);
      acc = WMMA_BF16(a.v, bf[kk].v, acc);
    }
#pragma unroll
    for (int r = 0; r < 8; r++) {
      float v = acc[r] + tb1c;
      atomicAdd(&score[mt * 16 + r + hi8], gelu_exact(v) * tw2c);
    }
  }
  __syncthreads();

  // softmax over T=256
  red[tid] = score[tid];
  __syncthreads();
  for (int off = 128; off > 0; off >>= 1) {
    if (tid < off) red[tid] = fmaxf(red[tid], red[tid + off]);
    __syncthreads();
  }
  float mx = red[0];
  __syncthreads();
  float e = __expf(score[tid] - mx);
  score[tid] = e;
  red[tid] = e;
  __syncthreads();
  for (int off = 128; off > 0; off >>= 1) {
    if (tid < off) red[tid] += red[tid + off];
    __syncthreads();
  }
  float denom = red[0];
  __syncthreads();

  // weighted sum: out[b][c] = sum_t attn[t] * xt[t][c]
  int cc = tid & 127, half = tid >> 7;
  float a2 = 0.0f;
  for (int t = half * 128; t < half * 128 + 128; t++)
    a2 += score[t] * bf2f(xbuf[t * HID + cc]);
  red[tid] = a2;
  __syncthreads();
  if (half == 0)
    out[(long)blockIdx.x * HID + cc] = (red[cc] + red[128 + cc]) / denom;
}

// ================= launch =================
extern "C" void kernel_launch(void* const* d_in, const int* in_sizes, int n_in,
                              void* d_out, int out_size, void* d_ws, size_t ws_size,
                              hipStream_t stream) {
  (void)in_sizes; (void)n_in; (void)out_size; (void)ws_size;
  char* ws = (char*)d_ws;
  const float* x = (const float*)d_in[0];
  float* xt = (float*)(ws + OFF_XT);

  (void)hipFuncSetAttribute((const void*)gcn_kernel,
                            hipFuncAttributeMaxDynamicSharedMemorySize, 64 * 1024);
  (void)hipFuncSetAttribute((const void*)attn_kernel,
                            hipFuncAttributeMaxDynamicSharedMemorySize, 72 * 1024);

  PackArgs pa;
  for (int i = 0; i < 26; i++) pa.p[i] = (const float*)d_in[i];
  pack_kernel<<<64, 256, 0, stream>>>(pa, ws);

  const int nBlocks = (64 * 256) / SAMP;                              // 8192
  const size_t smem2 = (size_t)(MROWS * HID * 2 + 128 * 128 * 2 + JP * 64 * 2); // 62 KB
  gcn_kernel<<<nBlocks, 256, smem2, stream>>>(x, ws, xt);

  const size_t smem3 = (size_t)(256 * 128 * 2 + 512 * 4);             // 66 KB
  attn_kernel<<<64, 256, smem3, stream>>>(xt, ws,
      (const float*)d_in[23], (const float*)d_in[24], (const float*)d_in[25],
      (float*)d_out);
}
